// MaskedMHCA_21543555957466
// MI455X (gfx1250) — matrix operand order repaired and checked by reading, not compiled, not run on hardware
//
#include <hip/hip_runtime.h>

#define Bn 2
#define Cn 512
#define Tn 2048
#define Hn 16
#define Dn 32
#define Nn (Bn*Tn)

typedef __attribute__((ext_vector_type(16))) __bf16 v16bf;
typedef __attribute__((ext_vector_type(8)))  float  v8f;

struct __attribute__((aligned(16))) U4 { unsigned x, y, z, w; };
struct __attribute__((aligned(16))) U8 { U4 lo, hi; };

// Load 16 bf16 as a WMMA fragment: elements 0..7 from p0, 8..15 from p1.
__device__ inline v16bf ldpair(const __bf16* p0, const __bf16* p1) {
    U8 t;
    t.lo = *(const U4*)p0;
    t.hi = *(const U4*)p1;
    return __builtin_bit_cast(v16bf, t);
}

__device__ inline v8f wmma_bf16(v16bf a, v16bf b, v8f c) {
    // (neg_a, A, neg_b, B, c_mod, C, reuse_a, reuse_b)
    return __builtin_amdgcn_wmma_f32_16x16x32_bf16(false, a, false, b, (short)0, c, false, false);
}

// ---------------------------------------------------------------------------
// K0: convert the four 512x512 weight matrices to bf16
// ---------------------------------------------------------------------------
__global__ __launch_bounds__(256) void mhca_cvt_w(
    const float* wq, const float* wk, const float* wv, const float* wp,
    __bf16* wqb, __bf16* wkb, __bf16* wvb, __bf16* wpb)
{
    int i = blockIdx.x * blockDim.x + threadIdx.x;
    if (i >= Cn * Cn) return;
    wqb[i] = (__bf16)wq[i];
    wkb[i] = (__bf16)wk[i];
    wvb[i] = (__bf16)wv[i];
    wpb[i] = (__bf16)wp[i];
}

// ---------------------------------------------------------------------------
// K1: fused dwconv3 (+mask) + channel LayerNorm for q,k,v. One block per (b,t),
// thread per channel. Output bf16 activations in [B,T,C] (channel-contiguous).
// ---------------------------------------------------------------------------
__device__ inline float blk_reduce_512(float v, float* sm) {
    #pragma unroll
    for (int o = 16; o; o >>= 1) v += __shfl_xor(v, o, 32);
    __syncthreads();                       // protect sm reuse across calls
    if ((threadIdx.x & 31) == 0) sm[threadIdx.x >> 5] = v;
    __syncthreads();
    float r = 0.f;
    #pragma unroll
    for (int i = 0; i < 16; ++i) r += sm[i];
    return r;
}

__global__ __launch_bounds__(512) void mhca_convln(
    const float* __restrict__ x1, const float* __restrict__ x2,
    const unsigned char* __restrict__ mk,
    const float* __restrict__ qw, const float* __restrict__ kw, const float* __restrict__ vw,
    const float* __restrict__ qlw, const float* __restrict__ qlb,
    const float* __restrict__ klw, const float* __restrict__ klb,
    const float* __restrict__ vlw, const float* __restrict__ vlb,
    __bf16* __restrict__ qact, __bf16* __restrict__ kact, __bf16* __restrict__ vact)
{
    __shared__ float sm[16];
    const int n = blockIdx.x;
    const int b = n >> 11;            // T = 2048
    const int t = n & (Tn - 1);
    const int c = threadIdx.x;
    const float mf = mk[n] ? 1.f : 0.f;

    const size_t xi = (size_t)b * Cn * Tn + (size_t)c * Tn + t;
    float l1 = (t > 0)      ? x1[xi - 1] : 0.f;
    float c1 = x1[xi];
    float r1 = (t < Tn - 1) ? x1[xi + 1] : 0.f;
    float l2 = (t > 0)      ? x2[xi - 1] : 0.f;
    float c2 = x2[xi];
    float r2 = (t < Tn - 1) ? x2[xi + 1] : 0.f;

    float qv = (qw[c*3+0]*l2 + qw[c*3+1]*c2 + qw[c*3+2]*r2) * mf;
    float kv = (kw[c*3+0]*l1 + kw[c*3+1]*c1 + kw[c*3+2]*r1) * mf;
    float vv = (vw[c*3+0]*l1 + vw[c*3+1]*c1 + vw[c*3+2]*r1) * mf;

    const float inv = 1.f / (float)Cn;
    float s, ss, mu, var;

    s  = blk_reduce_512(qv, sm);
    ss = blk_reduce_512(qv * qv, sm);
    mu = s * inv; var = ss * inv - mu * mu;
    qact[(size_t)n * Cn + c] = (__bf16)((qv - mu) * rsqrtf(var + 1e-5f) * qlw[c] + qlb[c]);

    s  = blk_reduce_512(kv, sm);
    ss = blk_reduce_512(kv * kv, sm);
    mu = s * inv; var = ss * inv - mu * mu;
    kact[(size_t)n * Cn + c] = (__bf16)((kv - mu) * rsqrtf(var + 1e-5f) * klw[c] + klb[c]);

    s  = blk_reduce_512(vv, sm);
    ss = blk_reduce_512(vv * vv, sm);
    mu = s * inv; var = ss * inv - mu * mu;
    vact[(size_t)n * Cn + c] = (__bf16)((vv - mu) * rsqrtf(var + 1e-5f) * vlw[c] + vlb[c]);
}

// ---------------------------------------------------------------------------
// K2: conv1x1 GEMMs for q,k,v (z selects matrix). out[n,o] = W[o,:] . act[n,:]
// One 16x16 tile per wave, 16 WMMAs over K=512.
// q,k stored bf16 [N,C]; v stored transposed bf16 [B,C,T] for attention PV.
// ---------------------------------------------------------------------------
__global__ __launch_bounds__(256) void mhca_gemm_qkv(
    const __bf16* __restrict__ qact, const __bf16* __restrict__ kact, const __bf16* __restrict__ vact,
    const __bf16* __restrict__ wqb, const __bf16* __restrict__ wkb, const __bf16* __restrict__ wvb,
    const float* __restrict__ bq, const float* __restrict__ bk, const float* __restrict__ bv,
    __bf16* __restrict__ qo, __bf16* __restrict__ ko, __bf16* __restrict__ vTo)
{
    const int z = blockIdx.z;
    const __bf16* act  = (z == 0) ? qact : (z == 1) ? kact : vact;
    const __bf16* W    = (z == 0) ? wqb  : (z == 1) ? wkb  : wvb;
    const float*  bias = (z == 0) ? bq   : (z == 1) ? bk   : bv;

    const int lane  = threadIdx.x & 31;
    const int wid   = threadIdx.x >> 5;
    const int laneN = lane & 15;
    const int half  = lane >> 4;

    const int o0 = (blockIdx.y * 8 + wid) * 16;
    const int n0 = blockIdx.x * 16;

    const __bf16* arow = W   + (size_t)(o0 + laneN) * Cn;   // A: M = o0+laneN
    const __bf16* brow = act + (size_t)(n0 + laneN) * Cn;   // B: N = n0+laneN

    v8f acc = {};
    #pragma unroll 4
    for (int k0 = 0; k0 < Cn; k0 += 32) {
        // A fragment: K = half*16 + e (16 contiguous)
        v16bf a = ldpair(arow + k0 + half * 16, arow + k0 + half * 16 + 8);
        // B fragment: K = half*16 + e (16 contiguous)
        v16bf bm = ldpair(brow + k0 + half * 16, brow + k0 + half * 16 + 8);
        acc = wmma_bf16(a, bm, acc);
    }

    const int n = n0 + laneN;
    if (z < 2) {
        __bf16* out = (z == 0) ? qo : ko;
        #pragma unroll
        for (int r = 0; r < 8; ++r) {
            int o = o0 + r + half * 8;
            out[(size_t)n * Cn + o] = (__bf16)(acc[r] + bias[o]);
        }
    } else {
        const int bb = n >> 11, t = n & (Tn - 1);
        #pragma unroll
        for (int r = 0; r < 8; ++r) {
            int o = o0 + r + half * 8;
            vTo[((size_t)bb * Cn + o) * Tn + t] = (__bf16)(acc[r] + bias[o]);
        }
    }
}

// ---------------------------------------------------------------------------
// K3: flash attention. One wave per 16-query tile per (b,h). 32 keys per step:
// 2 WMMAs (S), online softmax (f32), P via LDS C->A layout bounce, 2 WMMAs (PV).
// ---------------------------------------------------------------------------
__global__ __launch_bounds__(32) void mhca_attn(
    const __bf16* __restrict__ qbuf, const __bf16* __restrict__ kbuf,
    const __bf16* __restrict__ vT, const unsigned char* __restrict__ mk,
    __bf16* __restrict__ obuf)
{
    __shared__ __bf16 plds[16 * 32];
    const int b = blockIdx.z, h = blockIdx.y, tq0 = blockIdx.x * 16;
    const int lane  = threadIdx.x & 31;
    const int laneN = lane & 15;
    const int half  = lane >> 4;
    const float scale = 0.17677669529663687f;  // 1/sqrt(32)

    // Q as A-fragment (row = query, K = d), loaded once.
    const __bf16* qp = qbuf + (size_t)(b * Tn + tq0 + laneN) * Cn + h * Dn + half * 16;
    const v16bf qa = ldpair(qp, qp + 8);

    float mrow[8], lrow[8];
    v8f acc0 = {}, acc1 = {};
    #pragma unroll
    for (int r = 0; r < 8; ++r) { mrow[r] = -3.0e38f; lrow[r] = 0.f; }

    for (int j0 = 0; j0 < Tn; j0 += 32) {
        // K as B-fragments: N = key, K = d (contiguous in memory)
        const __bf16* kp0 = kbuf + (size_t)(b * Tn + j0 + laneN) * Cn + h * Dn + half * 16;
        const __bf16* kp1 = kp0 + (size_t)16 * Cn;
        v16bf kb0 = ldpair(kp0, kp0 + 8);
        v16bf kb1 = ldpair(kp1, kp1 + 8);
        v8f zc = {};
        v8f s0 = wmma_bf16(qa, kb0, zc);   // keys j0..j0+15
        v8f s1 = wmma_bf16(qa, kb1, zc);   // keys j0+16..j0+31

        const bool v0 = mk[b * Tn + j0 + laneN] != 0;
        const bool v1 = mk[b * Tn + j0 + 16 + laneN] != 0;

        #pragma unroll
        for (int r = 0; r < 8; ++r) {
            float a0 = v0 ? s0[r] * scale : -1.0e30f;
            float a1 = v1 ? s1[r] * scale : -1.0e30f;
            float tm = fmaxf(a0, a1);
            tm = fmaxf(tm, __shfl_xor(tm, 1, 32));
            tm = fmaxf(tm, __shfl_xor(tm, 2, 32));
            tm = fmaxf(tm, __shfl_xor(tm, 4, 32));
            tm = fmaxf(tm, __shfl_xor(tm, 8, 32));
            float mnew = fmaxf(mrow[r], tm);
            float corr = __expf(mrow[r] - mnew);
            float p0 = v0 ? __expf(a0 - mnew) : 0.f;
            float p1 = v1 ? __expf(a1 - mnew) : 0.f;
            float rs = p0 + p1;
            rs += __shfl_xor(rs, 1, 32);
            rs += __shfl_xor(rs, 2, 32);
            rs += __shfl_xor(rs, 4, 32);
            rs += __shfl_xor(rs, 8, 32);
            lrow[r] = lrow[r] * corr + rs;
            mrow[r] = mnew;
            acc0[r] *= corr;
            acc1[r] *= corr;
            // P in C-layout -> row-major 16x32 LDS tile
            plds[(r + half * 8) * 32 + laneN]      = (__bf16)p0;
            plds[(r + half * 8) * 32 + laneN + 16] = (__bf16)p1;
        }
        __syncthreads();
        // Re-read P as A-fragment (row = query, K = key-in-32)
        const __bf16* pp = &plds[laneN * 32 + half * 16];
        v16bf pa = ldpair(pp, pp + 8);
        // V as B-fragments from transposed [B,C,T]: N = d, K = key (contiguous)
        const __bf16* vp0 = vT + (size_t)(b * Cn + h * Dn + laneN) * Tn + j0 + half * 16;
        const __bf16* vp1 = vT + (size_t)(b * Cn + h * Dn + 16 + laneN) * Tn + j0 + half * 16;
        v16bf vb0 = ldpair(vp0, vp0 + 8);
        v16bf vb1 = ldpair(vp1, vp1 + 8);
        acc0 = wmma_bf16(pa, vb0, acc0);
        acc1 = wmma_bf16(pa, vb1, acc1);
        __syncthreads();
    }

    #pragma unroll
    for (int r = 0; r < 8; ++r) {
        float invl = lrow[r] > 0.f ? 1.f / lrow[r] : 0.f;
        int t = tq0 + r + half * 8;
        size_t base = (size_t)(b * Tn + t) * Cn + h * Dn;
        obuf[base + laneN]      = (__bf16)(acc0[r] * invl);
        obuf[base + 16 + laneN] = (__bf16)(acc1[r] * invl);
    }
}

// ---------------------------------------------------------------------------
// K4: final projection GEMM + bias + mask, fp32 output in [B,C,T]
// ---------------------------------------------------------------------------
__global__ __launch_bounds__(256) void mhca_proj(
    const __bf16* __restrict__ obuf, const __bf16* __restrict__ wpb,
    const float* __restrict__ bp, const unsigned char* __restrict__ mk,
    float* __restrict__ out)
{
    const int lane  = threadIdx.x & 31;
    const int wid   = threadIdx.x >> 5;
    const int laneN = lane & 15;
    const int half  = lane >> 4;
    const int o0 = (blockIdx.y * 8 + wid) * 16;
    const int n0 = blockIdx.x * 16;

    const __bf16* arow = wpb  + (size_t)(o0 + laneN) * Cn;
    const __bf16* brow = obuf + (size_t)(n0 + laneN) * Cn;

    v8f acc = {};
    #pragma unroll 4
    for (int k0 = 0; k0 < Cn; k0 += 32) {
        v16bf a  = ldpair(arow + k0 + half * 16, arow + k0 + half * 16 + 8);
        v16bf bm = ldpair(brow + k0 + half * 16, brow + k0 + half * 16 + 8);
        acc = wmma_bf16(a, bm, acc);
    }

    const int n = n0 + laneN;
    const int bb = n >> 11, t = n & (Tn - 1);
    const float mf = mk[n] ? 1.f : 0.f;
    #pragma unroll
    for (int r = 0; r < 8; ++r) {
        int o = o0 + r + half * 8;
        out[((size_t)bb * Cn + o) * Tn + t] = (acc[r] + bp[o]) * mf;
    }
}

// K5: append mask (second tuple output) as 0/1 floats
__global__ __launch_bounds__(256) void mhca_maskcopy(
    const unsigned char* __restrict__ mk, float* __restrict__ dst)
{
    int i = blockIdx.x * blockDim.x + threadIdx.x;
    if (i < Bn * Tn) dst[i] = mk[i] ? 1.f : 0.f;
}

// ---------------------------------------------------------------------------
extern "C" void kernel_launch(void* const* d_in, const int* in_sizes, int n_in,
                              void* d_out, int out_size, void* d_ws, size_t ws_size,
                              hipStream_t stream) {
    const float* x1 = (const float*)d_in[0];
    const float* x2 = (const float*)d_in[1];
    const unsigned char* mk = (const unsigned char*)d_in[2];
    const float* qw  = (const float*)d_in[3];
    const float* kw  = (const float*)d_in[4];
    const float* vw  = (const float*)d_in[5];
    const float* qlw = (const float*)d_in[6];
    const float* qlb = (const float*)d_in[7];
    const float* klw = (const float*)d_in[8];
    const float* klb = (const float*)d_in[9];
    const float* vlw = (const float*)d_in[10];
    const float* vlb = (const float*)d_in[11];
    const float* wq  = (const float*)d_in[12];
    const float* bq  = (const float*)d_in[13];
    const float* wk  = (const float*)d_in[14];
    const float* bk  = (const float*)d_in[15];
    const float* wv  = (const float*)d_in[16];
    const float* bv  = (const float*)d_in[17];
    const float* wp  = (const float*)d_in[18];
    const float* bp  = (const float*)d_in[19];

    char* w = (char*)d_ws;
    auto carve = [&](size_t bytes) {
        char* p = w;
        w += (bytes + 255) & ~(size_t)255;
        return (void*)p;
    };
    __bf16* wqb  = (__bf16*)carve((size_t)Cn * Cn * 2);
    __bf16* wkb  = (__bf16*)carve((size_t)Cn * Cn * 2);
    __bf16* wvb  = (__bf16*)carve((size_t)Cn * Cn * 2);
    __bf16* wpb  = (__bf16*)carve((size_t)Cn * Cn * 2);
    __bf16* qact = (__bf16*)carve((size_t)Nn * Cn * 2);
    __bf16* kact = (__bf16*)carve((size_t)Nn * Cn * 2);
    __bf16* vact = (__bf16*)carve((size_t)Nn * Cn * 2);
    __bf16* qbuf = (__bf16*)carve((size_t)Nn * Cn * 2);
    __bf16* kbuf = (__bf16*)carve((size_t)Nn * Cn * 2);
    __bf16* vTbf = (__bf16*)carve((size_t)Nn * Cn * 2);   // [B,C,T]
    __bf16* obuf = (__bf16*)carve((size_t)Nn * Cn * 2);

    mhca_cvt_w<<<(Cn * Cn + 255) / 256, 256, 0, stream>>>(
        wq, wk, wv, wp, wqb, wkb, wvb, wpb);

    mhca_convln<<<Nn, 512, 0, stream>>>(
        x1, x2, mk, qw, kw, vw, qlw, qlb, klw, klb, vlw, vlb, qact, kact, vact);

    mhca_gemm_qkv<<<dim3(Nn / 16, Cn / 16 / 8, 3), 256, 0, stream>>>(
        qact, kact, vact, wqb, wkb, wvb, bq, bk, bv, qbuf, kbuf, vTbf);

    mhca_attn<<<dim3(Tn / 16, Hn, Bn), 32, 0, stream>>>(
        qbuf, kbuf, vTbf, mk, obuf);

    mhca_proj<<<dim3(Nn / 16, Cn / 16 / 8), 256, 0, stream>>>(
        obuf, wpb, bp, mk, (float*)d_out);

    if (out_size >= Bn * Cn * Tn + Bn * Tn) {
        mhca_maskcopy<<<(Bn * Tn + 255) / 256, 256, 0, stream>>>(
            mk, (float*)d_out + (size_t)Bn * Cn * Tn);
    }
}
